// Classifier_89309549953249
// MI455X (gfx1250) — matrix operand order repaired and checked by reading, not compiled, not run on hardware
//
#include <hip/hip_runtime.h>
#include <hip/hip_bf16.h>

typedef __bf16 bf16_t;
typedef __attribute__((ext_vector_type(16))) __bf16 v16bf;
typedef __attribute__((ext_vector_type(8)))  __bf16 v8bf;
typedef __attribute__((ext_vector_type(8)))  float  v8f;

#define NEG_SLOPE 0.2f

// ---------------------------------------------------------------------------
// WMMA helpers (CDNA5 16x16x32 bf16 -> f32)
// ---------------------------------------------------------------------------
static __device__ __forceinline__ v8f wmma_bf16(v16bf a, v16bf b, v8f c) {
  return __builtin_amdgcn_wmma_f32_16x16x32_bf16(false, a, false, b, (short)0, c,
                                                 false, false);
}

// B operand: weights stored transposed bf16 [32][Kdim] in ws.
// Lane L holds column n = tN*16 + L%16, K = ks*32 + (L&16?8:0) + {0..7} and
// +16 + {0..7} (same as A) -> two 16B vector loads from global (L2-resident).
static __device__ __forceinline__ v16bf load_b(const bf16_t* Bt, int Kdim,
                                               int tN, int ks, int lane) {
  int n  = tN * 16 + (lane & 15);
  int kb = ks * 32 + ((lane & 16) ? 8 : 0);
  v8bf lo = *(const v8bf*)(Bt + n * Kdim + kb);
  v8bf hi = *(const v8bf*)(Bt + n * Kdim + kb + 16);
  return __builtin_shufflevector(lo, hi, 0, 1, 2, 3, 4, 5, 6, 7, 8, 9, 10, 11,
                                 12, 13, 14, 15);
}

// A operand from an LDS row (bf16): row = lane%16, two contiguous 8-elem runs
// K = ks*32 + half8 + {0..7} and +16 + {0..7}  (ISA 16-bit A 16x32 layout).
static __device__ __forceinline__ v16bf load_a_lds(const bf16_t* row, int ks,
                                                   int half8) {
  int b = ks * 32 + half8;
  v8bf lo = *(const v8bf*)(row + b);
  v8bf hi = *(const v8bf*)(row + b + 16);
  return __builtin_shufflevector(lo, hi, 0, 1, 2, 3, 4, 5, 6, 7, 8, 9, 10, 11,
                                 12, 13, 14, 15);
}

// A operand directly from a global f32 row (node GEMMs): same K pattern,
// converted f32 -> bf16 in registers (v_cvt_pk_bf16_f32 path).
static __device__ __forceinline__ v16bf cvt_a_row(const float* pr, int ks,
                                                  int half8) {
  int b = ks * 32 + half8;
  v16bf a;
#pragma unroll
  for (int j = 0; j < 8; ++j) a[j] = (__bf16)pr[b + j];
#pragma unroll
  for (int j = 0; j < 8; ++j) a[8 + j] = (__bf16)pr[b + 16 + j];
  return a;
}

// ---------------------------------------------------------------------------
// Weight pre-transpose f32[K][N] -> bf16[N][K] (Wc padded to 16 columns)
// ---------------------------------------------------------------------------
#define OFF_W1T 0
#define OFF_W2T 2048
#define OFF_WE1T 3072
#define OFF_WE2T 6144
#define OFF_WCT 9216

__global__ void convert_weights_kernel(const float* W1, const float* W2,
                                       const float* We1, const float* We2,
                                       const float* Wc, bf16_t* wb) {
  int t = blockIdx.x * blockDim.x + threadIdx.x;
  int stride = gridDim.x * blockDim.x;
  for (int i = t; i < 32 * 64; i += stride) {
    int n = i / 64, k = i % 64;
    wb[OFF_W1T + i] = (__bf16)W1[k * 32 + n];
  }
  for (int i = t; i < 32 * 32; i += stride) {
    int n = i / 32, k = i % 32;
    wb[OFF_W2T + i] = (__bf16)W2[k * 32 + n];
  }
  for (int i = t; i < 32 * 96; i += stride) {
    int n = i / 96, k = i % 96;
    wb[OFF_WE1T + i] = (__bf16)We1[k * 32 + n];
    wb[OFF_WE2T + i] = (__bf16)We2[k * 32 + n];
  }
  for (int i = t; i < 16 * 32; i += stride) {
    int n = i / 32, k = i % 32;
    wb[OFF_WCT + i] = (__bf16)((n < 10) ? Wc[k * 10 + n] : 0.0f);
  }
}

// ---------------------------------------------------------------------------
// Node GEMM: H[N,32] = X[N,KD] @ W[KD,32]   (one 16-row tile per wave)
// ---------------------------------------------------------------------------
template <int KD>
__global__ __launch_bounds__(256) void node_gemm_kernel(const float* X,
                                                        const bf16_t* Bt,
                                                        float* H, int ntiles,
                                                        int nN) {
  int lane = threadIdx.x & 31;
  int wave = threadIdx.x >> 5;
  int tile = blockIdx.x * (blockDim.x >> 5) + wave;
  if (tile >= ntiles) return;  // wave-uniform: EXEC stays all-ones for WMMA
  int row = tile * 16 + (lane & 15);
  int rowc = row < nN ? row : nN - 1;
  const float* Xr = X + (size_t)rowc * KD;
  int half8 = (lane & 16) ? 8 : 0;
  v8f c0 = {0, 0, 0, 0, 0, 0, 0, 0};
  v8f c1 = {0, 0, 0, 0, 0, 0, 0, 0};
#pragma unroll
  for (int ks = 0; ks < KD / 32; ++ks) {
    v16bf a = cvt_a_row(Xr, ks, half8);
    c0 = wmma_bf16(a, load_b(Bt, KD, 0, ks, lane), c0);
    c1 = wmma_bf16(a, load_b(Bt, KD, 1, ks, lane), c1);
  }
  int n0 = lane & 15;
#pragma unroll
  for (int r = 0; r < 8; ++r) {
    int M = tile * 16 + r + half8;
    if (M < nN) {
      H[(size_t)M * 32 + n0] = c0[r];
      H[(size_t)M * 32 + 16 + n0] = c1[r];
    }
  }
}

// ---------------------------------------------------------------------------
// Per-node attention logit projections: as = h.a_src, ad = h.a_dst
// ---------------------------------------------------------------------------
__global__ void alphas_kernel(const float* H, const float* a_s,
                              const float* a_d, float* as_, float* ad_,
                              int nN) {
  int i = blockIdx.x * blockDim.x + threadIdx.x;
  if (i >= nN) return;
  const float* r = H + (size_t)i * 32;
  float s = 0.f, d = 0.f;
#pragma unroll
  for (int f = 0; f < 32; ++f) {
    float v = r[f];
    s += v * a_s[f];
    d += v * a_d[f];
  }
  as_[i] = s;
  ad_[i] = d;
}

// ---------------------------------------------------------------------------
// Scatter-softmax aggregation (3 passes + init + bias). Self loops are the
// last nN items of the (nE+nN)-long virtual edge list.
// ---------------------------------------------------------------------------
__global__ void init_kernel(float* m, float* z, float* agg, int nN) {
  int i = blockIdx.x * blockDim.x + threadIdx.x;
  if (i < nN * 32) agg[i] = 0.0f;
  if (i < nN) {
    m[i] = -__builtin_inff();
    z[i] = 0.0f;
  }
}

static __device__ __forceinline__ float gat_logit(const int* src,
                                                  const int* dst,
                                                  const float* as_,
                                                  const float* ad_, int i,
                                                  int nE, int& d) {
  int s = (i < nE) ? src[i] : (i - nE);
  d = (i < nE) ? dst[i] : (i - nE);
  float v = as_[s] + ad_[d];
  return v > 0.0f ? v : NEG_SLOPE * v;
}

__global__ void pass1_max_kernel(const int* src, const int* dst,
                                 const float* as_, const float* ad_, float* m,
                                 int nE, int nN) {
  int i = blockIdx.x * blockDim.x + threadIdx.x;
  if (i >= nE + nN) return;
  int d;
  float v = gat_logit(src, dst, as_, ad_, i, nE, d);
  if (v >= 0.0f)
    atomicMax((int*)(m + d), __float_as_int(v));
  else
    atomicMin((unsigned int*)(m + d), __float_as_uint(v));
}

__global__ void pass2_exp_kernel(const int* src, const int* dst,
                                 const float* as_, const float* ad_,
                                 const float* m, float* z, float* p, int nE,
                                 int nN) {
  int i = blockIdx.x * blockDim.x + threadIdx.x;
  if (i >= nE + nN) return;
  int d;
  float v = gat_logit(src, dst, as_, ad_, i, nE, d);
  float pe = __expf(v - m[d]);
  p[i] = pe;
  atomicAdd(z + d, pe);
}

__global__ void pass3_agg_kernel(const int* src, const int* dst,
                                 const float* p, const float* z,
                                 const float* H, float* agg, int nE, int nN) {
  int idx = blockIdx.x * blockDim.x + threadIdx.x;
  int i = idx >> 5, f = idx & 31;
  if (i >= nE + nN) return;
  int s = (i < nE) ? src[i] : (i - nE);
  int d = (i < nE) ? dst[i] : (i - nE);
  float a = p[i] / z[d];
  atomicAdd(agg + (size_t)d * 32 + f, a * H[(size_t)s * 32 + f]);
}

__global__ void bias_kernel(const float* agg, const float* b, float* Hout,
                            int nN) {
  int i = blockIdx.x * blockDim.x + threadIdx.x;
  if (i < nN * 32) Hout[i] = agg[i] + b[i & 31];
}

// ---------------------------------------------------------------------------
// Fused edge pipeline: e1 = [h1[s],h1[d],eattr]@We1+be1 (kept in registers),
// e2 = [h2[s],h2[d],e1]@We2+be2, out = e2@Wc+bc.  One 16-edge tile per wave;
// e1/e2 restaged C-layout -> A-layout through the wave's private LDS slab.
// ---------------------------------------------------------------------------
__global__ __launch_bounds__(256) void edge_fused_kernel(
    const int* src, const int* dst, const float* eattr, const float* h1,
    const float* h2, const bf16_t* We1t, const float* be1, const bf16_t* We2t,
    const float* be2, const bf16_t* Wct, const float* bc, float* out, int nE) {
  // row stride 104 bf16 (208B = 52 banks): the 16 per-lane b128 reads tile all
  // 64 LDS banks with zero conflicts.
  __shared__ __attribute__((aligned(32))) bf16_t As[8][16][104];
  int lane = threadIdx.x & 31;
  int wave = threadIdx.x >> 5;
  int tile = blockIdx.x * 8 + wave;
  int e0 = tile * 16;
  if (e0 >= nE) return;  // wave-uniform
  bf16_t(*A)[104] = As[wave];
  int half8 = (lane & 16) ? 8 : 0;
  int n0 = lane & 15;

  int eIdx = e0 + n0;
  if (eIdx >= nE) eIdx = nE - 1;
  int myidx = (lane < 16) ? src[eIdx] : dst[eIdx];

  // ---- stage A1 = [h1[src] | h1[dst] | edge_attr]  (16 x 96 bf16) ----
#pragma unroll
  for (int i = 0; i < 16; ++i) {
    int sN = __shfl(myidx, i, 32);
    int dN = __shfl(myidx, 16 + i, 32);
    int e = e0 + i < nE ? e0 + i : nE - 1;
    A[i][lane] = (__bf16)h1[(size_t)sN * 32 + lane];
    A[i][32 + lane] = (__bf16)h1[(size_t)dN * 32 + lane];
    A[i][64 + lane] = (__bf16)eattr[(size_t)e * 32 + lane];
  }
  __builtin_amdgcn_wave_barrier();

  // ---- e1 = A1 @ We1 + be1 ----
  v8f e1c0 = {0, 0, 0, 0, 0, 0, 0, 0};
  v8f e1c1 = {0, 0, 0, 0, 0, 0, 0, 0};
#pragma unroll
  for (int ks = 0; ks < 3; ++ks) {
    v16bf a = load_a_lds(&A[n0][0], ks, half8);
    e1c0 = wmma_bf16(a, load_b(We1t, 96, 0, ks, lane), e1c0);
    e1c1 = wmma_bf16(a, load_b(We1t, 96, 1, ks, lane), e1c1);
  }
  {
    float bb0 = be1[n0], bb1 = be1[16 + n0];
#pragma unroll
    for (int r = 0; r < 8; ++r) {
      e1c0[r] += bb0;
      e1c1[r] += bb1;
    }
  }
  __builtin_amdgcn_wave_barrier();

  // ---- stage A2 = [h2[src] | h2[dst] | e1]  (e1 from C regs, no HBM trip) --
#pragma unroll
  for (int i = 0; i < 16; ++i) {
    int sN = __shfl(myidx, i, 32);
    int dN = __shfl(myidx, 16 + i, 32);
    A[i][lane] = (__bf16)h2[(size_t)sN * 32 + lane];
    A[i][32 + lane] = (__bf16)h2[(size_t)dN * 32 + lane];
  }
#pragma unroll
  for (int r = 0; r < 8; ++r) {
    int M = r + half8;  // C layout: VGPR r -> row M = r + 8*lanehi, col lane%16
    A[M][64 + n0] = (__bf16)e1c0[r];
    A[M][80 + n0] = (__bf16)e1c1[r];
  }
  __builtin_amdgcn_wave_barrier();

  // ---- e2 = A2 @ We2 + be2 ----
  v8f e2c0 = {0, 0, 0, 0, 0, 0, 0, 0};
  v8f e2c1 = {0, 0, 0, 0, 0, 0, 0, 0};
#pragma unroll
  for (int ks = 0; ks < 3; ++ks) {
    v16bf a = load_a_lds(&A[n0][0], ks, half8);
    e2c0 = wmma_bf16(a, load_b(We2t, 96, 0, ks, lane), e2c0);
    e2c1 = wmma_bf16(a, load_b(We2t, 96, 1, ks, lane), e2c1);
  }
  {
    float bb0 = be2[n0], bb1 = be2[16 + n0];
#pragma unroll
    for (int r = 0; r < 8; ++r) {
      e2c0[r] += bb0;
      e2c1[r] += bb1;
    }
  }
  __builtin_amdgcn_wave_barrier();

  // ---- classifier: restage e2 into cols 0..31, one WMMA with padded Wc ----
#pragma unroll
  for (int r = 0; r < 8; ++r) {
    int M = r + half8;
    A[M][n0] = (__bf16)e2c0[r];
    A[M][16 + n0] = (__bf16)e2c1[r];
  }
  __builtin_amdgcn_wave_barrier();
  v8f cc = {0, 0, 0, 0, 0, 0, 0, 0};
  {
    v16bf a = load_a_lds(&A[n0][0], 0, half8);
    cc = wmma_bf16(a, load_b(Wct, 32, 0, 0, lane), cc);
  }
  if (n0 < 10) {
    float bcn = bc[n0];
#pragma unroll
    for (int r = 0; r < 8; ++r) {
      int M = r + half8;
      if (e0 + M < nE) out[(size_t)(e0 + M) * 10 + n0] = cc[r] + bcn;
    }
  }
}

// ---------------------------------------------------------------------------
// Host-side orchestration
// ---------------------------------------------------------------------------
extern "C" void kernel_launch(void* const* d_in, const int* in_sizes, int n_in,
                              void* d_out, int out_size, void* d_ws,
                              size_t ws_size, hipStream_t stream) {
  const float* x = (const float*)d_in[0];
  const int* ei = (const int*)d_in[1];
  const float* eattr = (const float*)d_in[2];
  const float* W1 = (const float*)d_in[3];
  const float* asrc1 = (const float*)d_in[4];
  const float* adst1 = (const float*)d_in[5];
  const float* b1 = (const float*)d_in[6];
  const float* We1 = (const float*)d_in[7];
  const float* be1 = (const float*)d_in[8];
  const float* W2 = (const float*)d_in[9];
  const float* asrc2 = (const float*)d_in[10];
  const float* adst2 = (const float*)d_in[11];
  const float* b2 = (const float*)d_in[12];
  const float* We2 = (const float*)d_in[13];
  const float* be2 = (const float*)d_in[14];
  const float* Wc = (const float*)d_in[15];
  const float* bc = (const float*)d_in[16];
  float* out = (float*)d_out;

  const int nN = in_sizes[0] / 64;
  const int nE = in_sizes[1] / 2;
  const int* srcI = ei;
  const int* dstI = ei + nE;
  const int EN = nE + nN;

  // ---- workspace carve-up (~60 MB) ----
  float* f = (float*)d_ws;
  size_t o = 0;
  float* h_raw = f + o; o += (size_t)nN * 32;
  float* h1f   = f + o; o += (size_t)nN * 32;
  float* h2f   = f + o; o += (size_t)nN * 32;
  float* agg   = f + o; o += (size_t)nN * 32;
  float* asb   = f + o; o += (size_t)nN;
  float* adb   = f + o; o += (size_t)nN;
  float* mb    = f + o; o += (size_t)nN;
  float* zb    = f + o; o += (size_t)nN;
  float* pb    = f + o; o += (size_t)EN;
  o = (o + 63) & ~(size_t)63;  // 256B-align the bf16 weight region
  bf16_t* wb = (bf16_t*)(f + o);

  const int ntiles = (nN + 15) / 16;
  const dim3 blk(256);
  const int gN = (nN + 255) / 256;
  const int gN32 = (nN * 32 + 255) / 256;
  const int gEN = (EN + 255) / 256;
  const int gEN32 = (int)(((size_t)EN * 32 + 255) / 256);
  const int gTile = (ntiles + 7) / 8;
  const int gEdge = ((nE + 15) / 16 + 7) / 8;

  convert_weights_kernel<<<16, blk, 0, stream>>>(W1, W2, We1, We2, Wc, wb);

  // ---- GAT layer 1 ----
  node_gemm_kernel<64><<<gTile, blk, 0, stream>>>(x, wb + OFF_W1T, h_raw,
                                                  ntiles, nN);
  alphas_kernel<<<gN, blk, 0, stream>>>(h_raw, asrc1, adst1, asb, adb, nN);
  init_kernel<<<gN32, blk, 0, stream>>>(mb, zb, agg, nN);
  pass1_max_kernel<<<gEN, blk, 0, stream>>>(srcI, dstI, asb, adb, mb, nE, nN);
  pass2_exp_kernel<<<gEN, blk, 0, stream>>>(srcI, dstI, asb, adb, mb, zb, pb,
                                            nE, nN);
  pass3_agg_kernel<<<gEN32, blk, 0, stream>>>(srcI, dstI, pb, zb, h_raw, agg,
                                              nE, nN);
  bias_kernel<<<gN32, blk, 0, stream>>>(agg, b1, h1f, nN);

  // ---- GAT layer 2 ----
  node_gemm_kernel<32><<<gTile, blk, 0, stream>>>(h1f, wb + OFF_W2T, h_raw,
                                                  ntiles, nN);
  alphas_kernel<<<gN, blk, 0, stream>>>(h_raw, asrc2, adst2, asb, adb, nN);
  init_kernel<<<gN32, blk, 0, stream>>>(mb, zb, agg, nN);
  pass1_max_kernel<<<gEN, blk, 0, stream>>>(srcI, dstI, asb, adb, mb, nE, nN);
  pass2_exp_kernel<<<gEN, blk, 0, stream>>>(srcI, dstI, asb, adb, mb, zb, pb,
                                            nE, nN);
  pass3_agg_kernel<<<gEN32, blk, 0, stream>>>(srcI, dstI, pb, zb, h_raw, agg,
                                              nE, nN);
  bias_kernel<<<gN32, blk, 0, stream>>>(agg, b2, h2f, nN);

  // ---- fused edge MLP1 + MLP2 + classifier (e1 never leaves the WGP) ----
  edge_fused_kernel<<<gEdge, blk, 0, stream>>>(
      srcI, dstI, eattr, h1f, h2f, wb + OFF_WE1T, be1, wb + OFF_WE2T, be2,
      wb + OFF_WCT, bc, out, nE);
}
